// FullAttention_Complex_65481071394886
// MI455X (gfx1250) — matrix operand order repaired and checked by reading, not compiled, not run on hardware
//
#include <hip/hip_runtime.h>

// ---------------------------------------------------------------------------
// Complex causal attention (flash-attention style) for MI455X / gfx1250.
//   B=2, L=S=2048, H=8, E=D=64.
// Pre-pass: f32 -> f16 repack into d_ws:
//   qr,qi,kr,ki : [B,H,L,E] f16
//   vr,vi       : [B,H,E,S] f16  (transposed, so V B-fragments are contiguous)
// Main kernel: one wave32 per 16-row query tile, 32 keys per iteration,
// v_wmma_f32_16x16x32_f16 for all GEMMs. Softmax of sqrt(sr^2+si^2) is done
// WITHOUT max-subtraction (scores are bounded ~[0,8] after the 1/sqrt(E)
// scale, so exp() cannot overflow and softmax is shift-invariant), and the
// row-sum is computed on the matrix pipe as P x ones. The causal mask is
// active in exactly ONE trailing 32-key block per query tile, so the hot
// loop is completely mask-free.
// ---------------------------------------------------------------------------

typedef __attribute__((ext_vector_type(16))) _Float16 v16h;
typedef __attribute__((ext_vector_type(8)))  _Float16 v8h;
typedef __attribute__((ext_vector_type(8)))  float    v8f;

#define B_  2
#define L_  2048
#define H_  8
#define E_  64

static __device__ __forceinline__ v8f wmma_f16f32(v16h a, v16h b, v8f c) {
  // D = A(16x32 f16) * B(32x16 f16) + C(16x16 f32)
  return __builtin_amdgcn_wmma_f32_16x16x32_f16(
      /*neg_a=*/false, a, /*neg_b=*/false, b,
      /*c_mod=*/(short)0, c, /*reuse_a=*/false, /*reuse_b=*/false);
}

// B fragment (32x16): lane l holds col n=l&15, K = 16*(l>>4) + 0..15
// contiguous -> one 32-byte load per lane.
static __device__ __forceinline__ v16h ld_bfrag(const _Float16* p) {
  return *(const v16h*)p;
}

// A fragment (16x32): lane l holds row m=l&15; halves are two contiguous
// 8-half chunks: K = 16*(l>>4)+0..7 and K = 16*(l>>4)+8..15.
static __device__ __forceinline__ v16h ld_afrag(const _Float16* p0, const _Float16* p1) {
  v8h lo = *(const v8h*)p0;
  v8h up = *(const v8h*)p1;
  v16h r;
#pragma unroll
  for (int i = 0; i < 8; ++i) { r[i] = lo[i]; r[8 + i] = up[i]; }
  return r;
}

// ---------------------------------------------------------------------------
// Pre-pass 1: Q/K f32 [B,L,H,E] -> f16 [B,H,L,E]
// ---------------------------------------------------------------------------
__global__ __launch_bounds__(256) void cvt_qk(
    const float* __restrict__ qr, const float* __restrict__ qi,
    const float* __restrict__ kr, const float* __restrict__ ki,
    _Float16* __restrict__ ws)
{
  const size_t N = (size_t)B_ * H_ * L_ * E_;
  size_t idx = (size_t)blockIdx.x * blockDim.x + threadIdx.x;
  if (idx >= N) return;
  const int e = (int)(idx & (E_ - 1));
  size_t t = idx >> 6;
  const int h = (int)(t & (H_ - 1)); t >>= 3;
  const int l = (int)(t & (L_ - 1));
  const int b = (int)(t >> 11);
  const size_t oq = (((size_t)(b * H_ + h)) * L_ + l) * E_ + e;
  ws[0 * N + oq] = (_Float16)qr[idx];
  ws[1 * N + oq] = (_Float16)qi[idx];
  ws[2 * N + oq] = (_Float16)kr[idx];
  ws[3 * N + oq] = (_Float16)ki[idx];
}

// ---------------------------------------------------------------------------
// Pre-pass 2: V f32 [B,L,H,E] -> f16 transposed [B,H,E,S] (LDS-tiled)
// ---------------------------------------------------------------------------
__global__ __launch_bounds__(256) void cvt_v(
    const float* __restrict__ vr, const float* __restrict__ vi,
    _Float16* __restrict__ ws)
{
  __shared__ _Float16 tr[32][33];
  __shared__ _Float16 ti[32][33];
  const size_t N = (size_t)B_ * H_ * L_ * E_;
  _Float16* outr = ws + 4 * N;
  _Float16* outi = ws + 5 * N;

  const int l0 = blockIdx.x * 32;
  const int e0 = blockIdx.y * 32;
  const int bh = blockIdx.z;
  const int b = bh >> 3, h = bh & 7;
  const int tx = threadIdx.x & 31;
  const int ty = threadIdx.x >> 5;  // 0..7

#pragma unroll
  for (int k = 0; k < 4; ++k) {
    const int l = l0 + ty + 8 * k;
    const size_t in = (((size_t)b * L_ + l) * H_ + h) * E_ + e0 + tx;
    tr[ty + 8 * k][tx] = (_Float16)vr[in];
    ti[ty + 8 * k][tx] = (_Float16)vi[in];
  }
  __syncthreads();
#pragma unroll
  for (int k = 0; k < 4; ++k) {
    const int e = e0 + ty + 8 * k;
    const size_t oo = (((size_t)bh) * E_ + e) * L_ + l0 + tx;
    outr[oo] = tr[tx][ty + 8 * k];   // vt[e][l] = v[l][e]
    outi[oo] = ti[tx][ty + 8 * k];
  }
}

// ---------------------------------------------------------------------------
// One 32-key block: 16 score WMMAs, elementwise exp(|s|*scale), P staging,
// 9 AV/row-sum WMMAs. MASKED=true only for the single diagonal block.
// ---------------------------------------------------------------------------
template <bool MASKED>
static __device__ __forceinline__ void process_block(
    int j0, int q0, int ln, int hl,
    const _Float16* __restrict__ krp, const _Float16* __restrict__ kip,
    const _Float16* __restrict__ vrp, const _Float16* __restrict__ vip,
    const v16h (&aqr)[2], const v16h (&aqi)[2], const v16h (&aqin)[2],
    const v16h& ones, _Float16* __restrict__ pst,
    v8f (&accr)[4], v8f (&acci)[4], v8f& ssum, float cexp)
{
#pragma unroll
  for (int n = 0; n < 2; ++n) {
    const int k0 = j0 + (n << 4);
    const size_t krow = (size_t)(k0 + ln) * E_ + 16 * hl;
    const v16h bkr0 = ld_bfrag(krp + krow);
    const v16h bkr1 = ld_bfrag(krp + krow + 32);
    const v16h bki0 = ld_bfrag(kip + krow);
    const v16h bki1 = ld_bfrag(kip + krow + 32);
    v8f z = {};
    v8f sr = wmma_f16f32(aqr[0], bkr0, z);      // qr.kr
    sr = wmma_f16f32(aqr[1], bkr1, sr);
    sr = wmma_f16f32(aqin[0], bki0, sr);        // - qi.ki
    sr = wmma_f16f32(aqin[1], bki1, sr);
    v8f si = wmma_f16f32(aqr[0], bki0, z);      // qr.ki
    si = wmma_f16f32(aqr[1], bki1, si);
    si = wmma_f16f32(aqi[0], bkr0, si);         // + qi.kr
    si = wmma_f16f32(aqi[1], bkr1, si);

    const int key = k0 + ln;
#pragma unroll
    for (int v = 0; v < 8; ++v) {
      const float m2 = __builtin_fmaf(si[v], si[v], sr[v] * sr[v]);
      float p = __builtin_amdgcn_exp2f(__builtin_amdgcn_sqrtf(m2) * cexp);
      if (MASKED) {
        const int row = q0 + v + 8 * hl;
        p = (key <= row) ? p : 0.0f;            // causal mask (strict upper)
      }
      // stage P (C/D layout) into per-wave LDS, row-major 16x32
      pst[(v + 8 * hl) * 32 + 16 * n + ln] = (_Float16)p;
    }
  }

  // Intra-wave cross-lane LDS dependency: wait for DS stores to land.
  asm volatile("s_wait_dscnt 0x0" ::: "memory");

  // Reload P as an A fragment (16x32): row-sum + A.V updates on matrix pipe.
  const v16h pa = ld_afrag(pst + ln * 32 + 16 * hl,
                           pst + ln * 32 + 16 * hl + 8);
  ssum = wmma_f16f32(pa, ones, ssum);
#pragma unroll
  for (int d = 0; d < 4; ++d) {
    const size_t vo = (size_t)(16 * d + ln) * L_ + j0 + 16 * hl;
    accr[d] = wmma_f16f32(pa, ld_bfrag(vrp + vo), accr[d]);
    acci[d] = wmma_f16f32(pa, ld_bfrag(vip + vo), acci[d]);
  }
}

// ---------------------------------------------------------------------------
// Flash complex attention: 1 wave per 16-row Q tile; 8 waves / block.
// ---------------------------------------------------------------------------
__global__ __launch_bounds__(256) void flash_cattn(
    const _Float16* __restrict__ ws, float* __restrict__ out)
{
  const size_t N = (size_t)B_ * H_ * L_ * E_;
  const _Float16* qr = ws;
  const _Float16* qi = ws + N;
  const _Float16* kr = ws + 2 * N;
  const _Float16* ki = ws + 3 * N;
  const _Float16* vr = ws + 4 * N;   // [B,H,E,S]
  const _Float16* vi = ws + 5 * N;

  __shared__ __align__(32) _Float16 pstage[8][16 * 32];  // per-wave P staging

  const int wave = threadIdx.x >> 5;
  const int lane = threadIdx.x & 31;
  const int hl   = lane >> 4;        // lane half (0/1)
  const int ln   = lane & 15;

  const int wg = blockIdx.x * 8 + wave;   // 0..2047
  const int bh = wg >> 7;                 // 128 q-tiles per (b,h)
  const int qt = wg & 127;
  const int b  = bh >> 3;
  const int h  = bh & 7;
  const int q0 = qt << 4;

  const _Float16* qrp = qr + (size_t)bh * L_ * E_;
  const _Float16* qip = qi + (size_t)bh * L_ * E_;
  const _Float16* krp = kr + (size_t)bh * L_ * E_;
  const _Float16* kip = ki + (size_t)bh * L_ * E_;
  const _Float16* vrp = vr + (size_t)bh * E_ * L_;
  const _Float16* vip = vi + (size_t)bh * E_ * L_;
  _Float16* pst = &pstage[wave][0];

  // Q A-fragments: two 16x32 e-chunks each; keep a negated imag copy since
  // f16 WMMA NEG bits only allow C-negate (ISA 7.12 NEG table).
  v16h aqr[2], aqi[2], aqin[2];
#pragma unroll
  for (int c = 0; c < 2; ++c) {
    const size_t ro = (size_t)(q0 + ln) * E_ + 32 * c;
    aqr[c]  = ld_afrag(qrp + ro + 16 * hl, qrp + ro + 16 * hl + 8);
    aqi[c]  = ld_afrag(qip + ro + 16 * hl, qip + ro + 16 * hl + 8);
    aqin[c] = -aqi[c];
  }

  // Constant all-ones B fragment: ssum += P * ones gives per-row sums on the
  // matrix pipe (C/D tile is constant along n, so every lane holds the sums).
  v16h ones;
#pragma unroll
  for (int i = 0; i < 16; ++i) ones[i] = (_Float16)1.0f;

  v8f accr[4], acci[4], ssum;
#pragma unroll
  for (int d = 0; d < 4; ++d) { accr[d] = (v8f){}; acci[d] = (v8f){}; }
  ssum = (v8f){};

  // p = exp2( mag * (1/sqrt(E)) * log2(e) )
  const float cexp = 0.125f * 1.44269504088896f;

  // Blocks with j0+32 <= q0 are entirely below the diagonal: mask-free.
  const int nfull = q0 >> 5;
  for (int jt = 0; jt < nfull; ++jt) {
    process_block<false>(jt << 5, q0, ln, hl, krp, kip, vrp, vip,
                         aqr, aqi, aqin, ones, pst, accr, acci, ssum, cexp);
  }
  // Exactly one trailing block straddles the diagonal: masked.
  process_block<true>(nfull << 5, q0, ln, hl, krp, kip, vrp, vip,
                      aqr, aqi, aqin, ones, pst, accr, acci, ssum, cexp);

  // Epilogue: normalize and store packed (real, imag) -> [B,L,H,D,2]
  float inv[8];
#pragma unroll
  for (int v = 0; v < 8; ++v) inv[v] = 1.0f / ssum[v];
#pragma unroll
  for (int d = 0; d < 4; ++d) {
#pragma unroll
    for (int v = 0; v < 8; ++v) {
      const int row = q0 + v + 8 * hl;
      const int dim = 16 * d + ln;
      const size_t off = ((((size_t)b * L_ + row) * H_ + h) * E_ + dim) * 2;
      float2 o;
      o.x = accr[d][v] * inv[v];
      o.y = acci[d][v] * inv[v];
      *reinterpret_cast<float2*>(out + off) = o;
    }
  }
}

// ---------------------------------------------------------------------------
extern "C" void kernel_launch(void* const* d_in, const int* in_sizes, int n_in,
                              void* d_out, int out_size, void* d_ws, size_t ws_size,
                              hipStream_t stream) {
  (void)in_sizes; (void)n_in; (void)out_size; (void)ws_size;
  const float* qr = (const float*)d_in[0];
  const float* qi = (const float*)d_in[1];
  const float* kr = (const float*)d_in[2];
  const float* ki = (const float*)d_in[3];
  const float* vr = (const float*)d_in[4];
  const float* vi = (const float*)d_in[5];
  _Float16* ws = (_Float16*)d_ws;   // needs 6 * B*H*L*E * 2 bytes = 24 MB
  float* out = (float*)d_out;

  const size_t N = (size_t)B_ * H_ * L_ * E_;
  cvt_qk<<<dim3((unsigned)((N + 255) / 256)), 256, 0, stream>>>(qr, qi, kr, ki, ws);
  cvt_v<<<dim3(L_ / 32, E_ / 32, B_ * H_), 256, 0, stream>>>(vr, vi, ws);

  const int total_waves = B_ * H_ * (L_ / 16);   // 2048
  flash_cattn<<<dim3(total_waves / 8), 256, 0, stream>>>(ws, out);
}
